// TensorEncoderText_39943195853472
// MI455X (gfx1250) — matrix operand order repaired and checked by reading, not compiled, not run on hardware
//
#include <hip/hip_runtime.h>
#include <hip/hip_bf16.h>

// ---------------------------------------------------------------------------
// Problem constants
// ---------------------------------------------------------------------------
#define BATCH      16384
#define WORD_DIM   300
#define WORD_PAD   320     // 10 * 32
#define TD         100     // TENSOR_DIM
#define TD_PAD     112     // 7 * 16
#define CAT        200     // 2*TD
#define CATF_LD    208     // 13 * 16   (f32 cat, j-dim padding)
#define CATB_LD    224     // 7  * 32   (bf16 cat, i-dim padding)
#define HID        512
#define HID_KPAD   128     // 4 * 32  (K padding of TD for l1)
#define EMB        1024

typedef __attribute__((ext_vector_type(16))) __bf16 v16bf;
typedef __attribute__((ext_vector_type(8)))  __bf16 v8bf;
typedef __attribute__((ext_vector_type(8)))  float  v8f;

union AB16 { v16bf v; v8bf h[2]; };

// ---------------------------------------------------------------------------
// WMMA helpers (CDNA5 v_wmma_f32_16x16x32_bf16)
// ---------------------------------------------------------------------------
__device__ __forceinline__ v8f wmma_bf16(v16bf a, v16bf b, v8f c) {
    return __builtin_amdgcn_wmma_f32_16x16x32_bf16(
        /*neg_a=*/false, a, /*neg_b=*/false, b,
        /*c_mod=*/(short)0, c, /*reuse_a=*/false, /*reuse_b=*/false);
}

// A fragment 16x32 bf16, row-major source, leading dim ld (elements).
// lane<16 : row=lane,    K offsets {0..15}
// lane>=16: row=lane-16, K offsets {16..31}
__device__ __forceinline__ v16bf load_a_frag(const __bf16* base, int ld, int lane) {
    int r    = lane & 15;
    int koff = (lane < 16) ? 0 : 16;
    const __bf16* p = base + r * ld + koff;
    AB16 u;
    u.h[0] = *(const v8bf*)(p);
    u.h[1] = *(const v8bf*)(p + 8);
    return u.v;
}

// B fragment K=32 x N=16.  Memory holds Bm[n][k] with k contiguous (ld = row
// stride between n rows).  lane<16: N=lane, K 0..15 ; lane>=16: N=lane-16, K 16..31
__device__ __forceinline__ v16bf load_b_frag(const __bf16* base, int ld, int lane) {
    int n    = lane & 15;
    int koff = (lane < 16) ? 0 : 16;
    const __bf16* p = base + n * ld + koff;
    AB16 u;
    u.h[0] = *(const v8bf*)(p);
    u.h[1] = *(const v8bf*)(p + 8);
    return u.v;
}

__device__ __forceinline__ float leaky(float v) { return v >= 0.f ? v : 0.1f * v; }

// ---------------------------------------------------------------------------
// Fill / convert kernels
// ---------------------------------------------------------------------------
__global__ void k0_ebf(const int* __restrict__ x, const float* __restrict__ embW,
                       __bf16* __restrict__ ebf) {
    int idx = blockIdx.x * blockDim.x + threadIdx.x;      // over BATCH*2*WORD_PAD
    if (idx >= BATCH * 2 * WORD_PAD) return;
    int i  = idx % WORD_PAD;
    int bs = idx / WORD_PAD;
    float v = 0.f;
    if (i < WORD_DIM) {
        int b = bs >> 1, s = bs & 1;
        v = embW[(size_t)x[b * 2 + s] * WORD_DIM + i];
    }
    ebf[idx] = (__bf16)v;
}

__global__ void k0_pd(const float* __restrict__ pdW, __bf16* __restrict__ pdp) {
    int idx = blockIdx.x * blockDim.x + threadIdx.x;      // TD_PAD*WORD_PAD
    if (idx >= TD_PAD * WORD_PAD) return;
    int i = idx % WORD_PAD, t = idx / WORD_PAD;
    float v = (t < TD && i < WORD_DIM) ? pdW[t * WORD_DIM + i] : 0.f;
    pdp[idx] = (__bf16)v;
}

__global__ void k0_T(const float* __restrict__ compT, __bf16* __restrict__ Tp) {
    int idx = blockIdx.x * blockDim.x + threadIdx.x;      // 100*208*224
    if (idx >= TD * CATF_LD * CATB_LD) return;
    int i = idx % CATB_LD;
    int j = (idx / CATB_LD) % CATF_LD;
    int k = idx / (CATB_LD * CATF_LD);
    float v = (j < CAT && i < CAT)
            ? compT[((size_t)k * CAT + j) * CAT + i] : 0.f;
    Tp[idx] = (__bf16)v;
}

__global__ void k0_cl(const float* __restrict__ clW, __bf16* __restrict__ clp) {
    int idx = blockIdx.x * blockDim.x + threadIdx.x;      // TD_PAD*CATB_LD
    if (idx >= TD_PAD * CATB_LD) return;
    int i = idx % CATB_LD, k = idx / CATB_LD;
    float v = (k < TD && i < CAT) ? clW[k * CAT + i] : 0.f;
    clp[idx] = (__bf16)v;
}

__global__ void k0_l1(const float* __restrict__ l1W, __bf16* __restrict__ l1p) {
    int idx = blockIdx.x * blockDim.x + threadIdx.x;      // HID*HID_KPAD
    if (idx >= HID * HID_KPAD) return;
    int i = idx % HID_KPAD, h = idx / HID_KPAD;
    float v = (i < TD) ? l1W[h * TD + i] : 0.f;
    l1p[idx] = (__bf16)v;
}

__global__ void k0_l2(const float* __restrict__ l2W, __bf16* __restrict__ l2p) {
    int idx = blockIdx.x * blockDim.x + threadIdx.x;      // EMB*HID
    if (idx >= EMB * HID) return;
    l2p[idx] = (__bf16)l2W[idx];
}

// Zero the padding columns of catf [B][208] (cols 200..207) and
// catbf [B][224] (cols 200..223).
__global__ void k0_catpad(float* __restrict__ catf, __bf16* __restrict__ catbf) {
    int idx = blockIdx.x * blockDim.x + threadIdx.x;      // BATCH*32
    if (idx >= BATCH * 32) return;
    int q = idx & 31, b = idx >> 5;
    if (q < 8)  catf[(size_t)b * CATF_LD + CAT + q] = 0.f;
    if (q < 24) catbf[(size_t)b * CATB_LD + CAT + q] = (__bf16)0.f;
}

// ---------------------------------------------------------------------------
// K1: projdown + leaky  ->  cat (f32 and bf16, padded)
//   grid (BATCH/16, 2), block 32 (one wave)
// ---------------------------------------------------------------------------
__global__ void __launch_bounds__(32)
k1_projdown(const __bf16* __restrict__ ebf,   // [B][2][320]
            const __bf16* __restrict__ pdp,   // [112][320]
            const float*  __restrict__ pdb,   // [100]
            float*  __restrict__ catf,        // [B][208]
            __bf16* __restrict__ catbf)       // [B][224]
{
    int lane = threadIdx.x & 31;
    int b0   = blockIdx.x * 16;
    int s    = blockIdx.y;

    v16bf A[10];
#pragma unroll
    for (int t = 0; t < 10; ++t)
        A[t] = load_a_frag(ebf + ((size_t)b0 * 2 + s) * WORD_PAD + 32 * t,
                           2 * WORD_PAD, lane);

    int n = lane & 15, mhi = (lane >> 4) * 8;
#pragma unroll
    for (int nt = 0; nt < 7; ++nt) {
        int t0 = nt * 16;
        v8f C = {};
#pragma unroll
        for (int t = 0; t < 10; ++t)
            C = wmma_bf16(A[t], load_b_frag(pdp + t0 * WORD_PAD + 32 * t,
                                            WORD_PAD, lane), C);
        int col = t0 + n;
        if (col < TD) {
            float bias = pdb[col];
#pragma unroll
            for (int r = 0; r < 8; ++r) {
                float v = leaky(C[r] + bias);
                int row = b0 + mhi + r;
                catf [(size_t)row * CATF_LD + s * TD + col] = v;
                catbf[(size_t)row * CATB_LD + s * TD + col] = (__bf16)v;
            }
        }
    }
}

// ---------------------------------------------------------------------------
// K1c: term2 = cat @ cl_W^T + cl_b   -> staged into comp[B][100]
//   grid BATCH/16, block 32
// ---------------------------------------------------------------------------
__global__ void __launch_bounds__(32)
k1c_term2(const __bf16* __restrict__ catbf,   // [B][224]
          const __bf16* __restrict__ clp,     // [112][224]
          const float*  __restrict__ clb,     // [100]
          float* __restrict__ comp)           // [B][100]
{
    int lane = threadIdx.x & 31;
    int b0   = blockIdx.x * 16;

    v16bf A[7];
#pragma unroll
    for (int t = 0; t < 7; ++t)
        A[t] = load_a_frag(catbf + (size_t)b0 * CATB_LD + 32 * t, CATB_LD, lane);

    int n = lane & 15, mhi = (lane >> 4) * 8;
#pragma unroll
    for (int nt = 0; nt < 7; ++nt) {
        int k0 = nt * 16;
        v8f C = {};
#pragma unroll
        for (int t = 0; t < 7; ++t)
            C = wmma_bf16(A[t], load_b_frag(clp + k0 * CATB_LD + 32 * t,
                                            CATB_LD, lane), C);
        int col = k0 + n;
        if (col < TD) {
            float bias = clb[col];
#pragma unroll
            for (int r = 0; r < 8; ++r)
                comp[(size_t)(b0 + mhi + r) * TD + col] = C[r] + bias;
        }
    }
}

// ---------------------------------------------------------------------------
// K2: term1[b,k] = cat_b^T T_k cat_b  (via V = cat * T_k^T, then rowdot cat)
//   then comp = leaky(term1 + term2)
//   grid (BATCH/64, 25), block 256 (8 waves).
//   Wave w: k = by*4 + (w&3), batch subtile-pair = (w>>2)*2.
//   2 subtiles per wave keeps A fragments at 112 VGPRs (no spill risk).
// ---------------------------------------------------------------------------
#define K2_BTILE 64
__global__ void __launch_bounds__(256)
k2_trilinear(const __bf16* __restrict__ catbf,  // [B][224]
             const float*  __restrict__ catf,   // [B][208]
             const __bf16* __restrict__ Tp,     // [100][208][224]
             float* __restrict__ comp)          // [B][100] (term2 on entry)
{
    int wave = threadIdx.x >> 5;                // 0..7
    int lane = threadIdx.x & 31;
    int k    = blockIdx.y * 4 + (wave & 3);     // < 100
    int sp   = (wave >> 2) * 2;                 // subtile pair base (0 or 2)
    int b0   = blockIdx.x * K2_BTILE;
    int bs   = b0 + 16 * sp;                    // this wave's first row

    __shared__ float lcat[K2_BTILE][CATF_LD];   // 53 KB
    for (int idx = threadIdx.x; idx < K2_BTILE * CATF_LD; idx += 256) {
        int r = idx / CATF_LD, c = idx % CATF_LD;
        lcat[r][c] = catf[(size_t)(b0 + r) * CATF_LD + c];
    }
    __syncthreads();

    // A fragments: 2 batch subtiles x 7 K-steps, held in registers (112 VGPRs)
    v16bf A[2][7];
#pragma unroll
    for (int s = 0; s < 2; ++s)
#pragma unroll
        for (int t = 0; t < 7; ++t)
            A[s][t] = load_a_frag(catbf + (size_t)(bs + 16 * s) * CATB_LD + 32 * t,
                                  CATB_LD, lane);

    const __bf16* Tk = Tp + (size_t)k * CATF_LD * CATB_LD;

    float acc0[8], acc1[8];
#pragma unroll
    for (int r = 0; r < 8; ++r) { acc0[r] = 0.f; acc1[r] = 0.f; }

    int n = lane & 15, mhi = (lane >> 4) * 8;

    for (int jt = 0; jt < 13; ++jt) {
        v8f C0 = {}, C1 = {};
#pragma unroll
        for (int t = 0; t < 7; ++t) {
            v16bf Bt = load_b_frag(Tk + (size_t)(16 * jt) * CATB_LD + 32 * t,
                                   CATB_LD, lane);
            C0 = wmma_bf16(A[0][t], Bt, C0);
            C1 = wmma_bf16(A[1][t], Bt, C1);
        }
        // rowdot with cat:  acc[s][r] += V[m, j] * cat[m, j]
        int jc = 16 * jt + n;
#pragma unroll
        for (int r = 0; r < 8; ++r) {
            acc0[r] += C0[r] * lcat[16 * sp      + mhi + r][jc];
            acc1[r] += C1[r] * lcat[16 * sp + 16 + mhi + r][jc];
        }
    }

    // reduce over the 16 N-lanes within each half-wave
#pragma unroll
    for (int r = 0; r < 8; ++r) {
        float v0 = acc0[r], v1 = acc1[r];
        v0 += __shfl_xor(v0, 1, 16);  v1 += __shfl_xor(v1, 1, 16);
        v0 += __shfl_xor(v0, 2, 16);  v1 += __shfl_xor(v1, 2, 16);
        v0 += __shfl_xor(v0, 4, 16);  v1 += __shfl_xor(v1, 4, 16);
        v0 += __shfl_xor(v0, 8, 16);  v1 += __shfl_xor(v1, 8, 16);
        acc0[r] = v0;  acc1[r] = v1;
    }

    if ((lane & 15) == 0) {
#pragma unroll
        for (int r = 0; r < 8; ++r) {
            int row0 = bs      + mhi + r;
            int row1 = bs + 16 + mhi + r;
            float v0 = acc0[r] + comp[(size_t)row0 * TD + k];   // + term2
            float v1 = acc1[r] + comp[(size_t)row1 * TD + k];
            comp[(size_t)row0 * TD + k] = leaky(v0);
            comp[(size_t)row1 * TD + k] = leaky(v1);
        }
    }
}

// ---------------------------------------------------------------------------
// K3: out = leaky(comp @ l1^T + b1) @ l2^T + b2 ; row L2-normalize
//   grid BATCH/16, block 256 (8 waves)
// ---------------------------------------------------------------------------
__global__ void __launch_bounds__(256)
k3_mlp(const float*  __restrict__ comp,   // [B][100]
       const __bf16* __restrict__ l1p,    // [512][128]
       const float*  __restrict__ l1b,    // [512]
       const __bf16* __restrict__ l2p,    // [1024][512]
       const float*  __restrict__ l2b,    // [1024]
       float* __restrict__ out)           // [B][1024]
{
    __shared__ __bf16 a1[16][HID_KPAD];   // 4 KB
    __shared__ __bf16 hid[16][HID];       // 16 KB
    __shared__ float  normacc[16];

    int b0   = blockIdx.x * 16;
    int wave = threadIdx.x >> 5;
    int lane = threadIdx.x & 31;

    for (int idx = threadIdx.x; idx < 16 * HID_KPAD; idx += 256) {
        int m = idx >> 7, i = idx & 127;
        a1[m][i] = (i < TD) ? (__bf16)comp[(size_t)(b0 + m) * TD + i]
                            : (__bf16)0.f;
    }
    if (threadIdx.x < 16) normacc[threadIdx.x] = 0.f;
    __syncthreads();

    int n = lane & 15, mhi = (lane >> 4) * 8;

    // ---- stage 1: hidden = leaky(comp @ l1^T + b1) ----
    v16bf A1[4];
#pragma unroll
    for (int t = 0; t < 4; ++t)
        A1[t] = load_a_frag(&a1[0][0] + 32 * t, HID_KPAD, lane);

#pragma unroll
    for (int nt = 0; nt < 4; ++nt) {
        int h0 = (wave * 4 + nt) * 16;
        v8f C = {};
#pragma unroll
        for (int t = 0; t < 4; ++t)
            C = wmma_bf16(A1[t], load_b_frag(l1p + (size_t)h0 * HID_KPAD + 32 * t,
                                             HID_KPAD, lane), C);
        float bias = l1b[h0 + n];
#pragma unroll
        for (int r = 0; r < 8; ++r)
            hid[mhi + r][h0 + n] = (__bf16)leaky(C[r] + bias);
    }
    __syncthreads();

    // ---- stage 2: out = hidden @ l2^T + b2, accumulate sum of squares ----
    float D[8][8];
    float sq[8];
#pragma unroll
    for (int r = 0; r < 8; ++r) sq[r] = 0.f;

#pragma unroll
    for (int nt = 0; nt < 8; ++nt) {
        int e0 = (wave * 8 + nt) * 16;
        v8f C = {};
#pragma unroll
        for (int t = 0; t < 16; ++t) {
            v16bf Af = load_a_frag(&hid[0][0] + 32 * t, HID, lane);
            v16bf Bt = load_b_frag(l2p + (size_t)e0 * HID + 32 * t, HID, lane);
            C = wmma_bf16(Af, Bt, C);
        }
        float bias = l2b[e0 + n];
#pragma unroll
        for (int r = 0; r < 8; ++r) {
            float v = C[r] + bias;
            D[nt][r] = v;
            sq[r] += v * v;
        }
    }

    // cross-lane (N) reduce of squared sums, then combine across waves in LDS
#pragma unroll
    for (int r = 0; r < 8; ++r) {
        float v = sq[r];
        v += __shfl_xor(v, 1, 16);
        v += __shfl_xor(v, 2, 16);
        v += __shfl_xor(v, 4, 16);
        v += __shfl_xor(v, 8, 16);
        sq[r] = v;
    }
    if ((lane & 15) == 0) {
#pragma unroll
        for (int r = 0; r < 8; ++r)
            atomicAdd(&normacc[mhi + r], sq[r]);
    }
    __syncthreads();

    float rn[8];
#pragma unroll
    for (int r = 0; r < 8; ++r)
        rn[r] = 1.0f / sqrtf(normacc[mhi + r]);

#pragma unroll
    for (int nt = 0; nt < 8; ++nt) {
        int e0 = (wave * 8 + nt) * 16;
#pragma unroll
        for (int r = 0; r < 8; ++r)
            out[(size_t)(b0 + mhi + r) * EMB + e0 + n] = D[nt][r] * rn[r];
    }
}

// ---------------------------------------------------------------------------
// Launch
// ---------------------------------------------------------------------------
extern "C" void kernel_launch(void* const* d_in, const int* in_sizes, int n_in,
                              void* d_out, int out_size, void* d_ws, size_t ws_size,
                              hipStream_t stream) {
    const int*   x     = (const int*)  d_in[0];
    // d_in[1] = lengths (unused)
    const float* embW  = (const float*)d_in[2];
    const float* pdW   = (const float*)d_in[3];
    const float* pdb   = (const float*)d_in[4];
    const float* compT = (const float*)d_in[5];
    const float* clW   = (const float*)d_in[6];
    const float* clb   = (const float*)d_in[7];
    const float* l1W   = (const float*)d_in[8];
    const float* l1b   = (const float*)d_in[9];
    const float* l2W   = (const float*)d_in[10];
    const float* l2b   = (const float*)d_in[11];
    float* out = (float*)d_out;

    char* ws = (char*)d_ws;
    size_t off = 0;
    auto alloc = [&](size_t bytes) { char* p = ws + off; off += (bytes + 255) & ~(size_t)255; return p; };

    __bf16* ebf   = (__bf16*)alloc((size_t)BATCH * 2 * WORD_PAD * 2);   // 20.97 MB
    float*  comp  = (float*)ebf;  // aliased: comp written after ebf consumed
    __bf16* pdp   = (__bf16*)alloc((size_t)TD_PAD * WORD_PAD * 2);
    __bf16* Tp    = (__bf16*)alloc((size_t)TD * CATF_LD * CATB_LD * 2); // 9.32 MB
    __bf16* clp   = (__bf16*)alloc((size_t)TD_PAD * CATB_LD * 2);
    __bf16* l1p   = (__bf16*)alloc((size_t)HID * HID_KPAD * 2);
    __bf16* l2p   = (__bf16*)alloc((size_t)EMB * HID * 2);              // 1.05 MB
    float*  catf  = (float*) alloc((size_t)BATCH * CATF_LD * 4);        // 13.63 MB
    __bf16* catbf = (__bf16*)alloc((size_t)BATCH * CATB_LD * 2);        // 7.34 MB
    (void)ws_size; (void)in_sizes; (void)n_in; (void)out_size;

    // ---- fills / conversions ----
    {
        int tot = BATCH * 2 * WORD_PAD;
        k0_ebf<<<(tot + 255) / 256, 256, 0, stream>>>(x, embW, ebf);
    }
    {
        int tot = TD_PAD * WORD_PAD;
        k0_pd<<<(tot + 255) / 256, 256, 0, stream>>>(pdW, pdp);
    }
    {
        int tot = TD * CATF_LD * CATB_LD;
        k0_T<<<(tot + 255) / 256, 256, 0, stream>>>(compT, Tp);
    }
    {
        int tot = TD_PAD * CATB_LD;
        k0_cl<<<(tot + 255) / 256, 256, 0, stream>>>(clW, clp);
    }
    {
        int tot = HID * HID_KPAD;
        k0_l1<<<(tot + 255) / 256, 256, 0, stream>>>(l1W, l1p);
    }
    {
        int tot = EMB * HID;
        k0_l2<<<(tot + 255) / 256, 256, 0, stream>>>(l2W, l2p);
    }
    {
        int tot = BATCH * 32;
        k0_catpad<<<(tot + 255) / 256, 256, 0, stream>>>(catf, catbf);
    }

    // ---- K1: projdown ----
    k1_projdown<<<dim3(BATCH / 16, 2), 32, 0, stream>>>(ebf, pdp, pdb, catf, catbf);

    // ---- K1c: term2 -> comp (aliased over ebf; ebf fully consumed by K1) ----
    k1c_term2<<<BATCH / 16, 32, 0, stream>>>(catbf, clp, clb, comp);

    // ---- K2: trilinear term (dominant) ----
    k2_trilinear<<<dim3(BATCH / K2_BTILE, 25), 256, 0, stream>>>(catbf, catf, Tp, comp);

    // ---- K3: MLP + L2 normalize ----
    k3_mlp<<<BATCH / 16, 256, 0, stream>>>(comp, l1p, l1b, l2p, l2b, out);
}
